// BranchNet1d_selfAttentionv1_73615739453982
// MI455X (gfx1250) — matrix operand order repaired and checked by reading, not compiled, not run on hardware
//
#include <hip/hip_runtime.h>
#include <stdint.h>

#define NXX    2048
#define BATCH  16
#define NMODE  16
#define WID    64
#define DMODEL 128

typedef __attribute__((ext_vector_type(16))) _Float16 v16h;
typedef __attribute__((ext_vector_type(8)))  _Float16 v8h;
typedef __attribute__((ext_vector_type(8)))  float    v8f;
typedef __attribute__((ext_vector_type(4)))  unsigned int u32x4;
typedef __attribute__((ext_vector_type(8)))  unsigned int u32x8;

// ---------------------------------------------------------------------------
// WMMA fragment helpers (CDNA5 wave32, v_wmma_f32_16x16x32_f16)
// A (16x32 f16): lane m = lane&15, kb = (lane>>4)*16; holds K=kb..kb+15 for row m
// B (32x16 f16): lane n = lane&15, kb = (lane>>4)*16; holds K=kb..kb+15 for col n
// C (16x16 f32): 8 VGPRs; lanes 0-15 rows 0-7, lanes 16-31 rows 8-15; col = lane&15
// Every operand load below is 16B-aligned contiguous -> 2 x b128 per lane.
// ---------------------------------------------------------------------------
__device__ __forceinline__ v16h join8(v8h lo, v8h hi) {
  return __builtin_shufflevector(lo, hi, 0, 1, 2, 3, 4, 5, 6, 7,
                                 8, 9, 10, 11, 12, 13, 14, 15);
}

__device__ __forceinline__ v16h load_a_frag(const _Float16* A, int lda) {
  const int lane = threadIdx.x & 31;
  const int m = lane & 15;
  const int kb = (lane >> 4) << 4;
  const _Float16* p = A + (long long)m * lda + kb;
  return join8(*(const v8h*)p, *(const v8h*)(p + 8));
}

// B fragment from a transposed source: B[k][n] = src[n*ld + k]  (contiguous)
__device__ __forceinline__ v16h load_b_t(const _Float16* src, int ld) {
  const int lane = threadIdx.x & 31;
  const int n = lane & 15;
  const int kb = (lane >> 4) << 4;
  const _Float16* p = src + (long long)n * ld + kb;
  return join8(*(const v8h*)p, *(const v8h*)(p + 8));
}

// B fragment from fragment-linear packed storage
__device__ __forceinline__ v16h load_b_packed(const _Float16* Bp, int kt, int nt,
                                              int ntiles) {
  const int lane = threadIdx.x & 31;
  const _Float16* p = Bp + (((long long)(kt * ntiles + nt) * 32 + lane) << 4);
  return join8(*(const v8h*)p, *(const v8h*)(p + 8));
}

// fragment-linear index for element (k, n) of a [K, N] B matrix
__device__ __forceinline__ long long fragidx(int k, int n, int ntiles) {
  return ((((long long)(k >> 5) * ntiles + (n >> 4)) * 2 + ((k >> 4) & 1)) * 16 +
          (n & 15)) * 16 + (k & 15);
}

__device__ __forceinline__ v8f wmma16(v16h a, v16h b, v8f c) {
  return __builtin_amdgcn_wmma_f32_16x16x32_f16(false, a, false, b, (short)0, c,
                                                false, false);
}

__device__ __forceinline__ float gelu_exact(float v) {
  return 0.5f * v * (1.0f + erff(v * 0.7071067811865475f));
}

// ---------------------------------------------------------------------------
// TDM: async 2-D tile load global -> LDS (Tensor Data Mover, gfx1250).
// D# groups 0/1 per CDNA5 ISA; 2-D tile, tensor dims ==
// tile dims, data_size = 2 bytes; wave-uniform operands in SGPR tuples.
// ---------------------------------------------------------------------------
__device__ __forceinline__ void tdm_load_2d(unsigned lds_off, const _Float16* gptr,
                                            unsigned tile0, unsigned tile1,
                                            unsigned long long stride0) {
  const unsigned long long ga = (unsigned long long)(uintptr_t)gptr;
  u32x4 g0;
  g0[0] = 1u;                                     // count = 1 (valid descriptor)
  g0[1] = lds_off;                                // lds_addr
  g0[2] = (unsigned)ga;                           // global_addr[31:0]
  g0[3] = (unsigned)(ga >> 32) | (2u << 30);      // global_addr[56:32] | type=2
  u32x8 g1;
  g1[0] = 1u << 16;                               // data_size = 1 (2 bytes)
  g1[1] = (tile0 & 0xFFFFu) << 16;                // tensor_dim0[15:0]
  g1[2] = (tile0 >> 16) | ((tile1 & 0xFFFFu) << 16);  // td0 hi | tensor_dim1 lo
  g1[3] = (tile1 >> 16) | (tile0 << 16);          // td1 hi | tile_dim0
  g1[4] = tile1 & 0xFFFFu;                        // tile_dim1 (tile_dim2 = 0)
  g1[5] = (unsigned)stride0;                      // tensor_dim0_stride[31:0]
  g1[6] = (unsigned)(stride0 >> 32) & 0xFFFFu;    // stride0[47:32]
  g1[7] = 0u;
  asm volatile("tensor_load_to_lds %0, %1" ::"s"(g0), "s"(g1) : "memory");
}

// ---------------------------------------------------------------------------
// Unified WMMA GEMM: D = A(f16,row-major)[M,K] * Bp(f16,fragment-packed)[K,N]
// (+Cinit)(+bias)(gelu?) -> f32 / f16 / transposed-f16 outputs.
// grid = (N/16, M/16, batch); block = 32 (one wave per 16x16 tile).
// ---------------------------------------------------------------------------
__global__ __launch_bounds__(32)
void wmma_gemm(const _Float16* __restrict__ A, const _Float16* __restrict__ Bp,
               const float* __restrict__ Cinit, float* __restrict__ Cf32,
               _Float16* __restrict__ Cf16, _Float16* __restrict__ CT,
               const float* __restrict__ bias, int gelu, int K, int lda, int ldc,
               long long sA, long long sB, long long sC) {
  const int tn = blockIdx.x, tm = blockIdx.y, bz = blockIdx.z;
  const int ntiles = gridDim.x;
  const int lane = threadIdx.x & 31;
  const int hi = lane >> 4, nn = lane & 15;
  const _Float16* Ab = A + (long long)bz * sA + (long long)tm * 16 * lda;
  const _Float16* Bb = Bp + (long long)bz * sB;
  const int row0 = tm * 16 + hi * 8;
  const int col = tn * 16 + nn;

  v8f acc;
  if (Cinit) {
#pragma unroll
    for (int r = 0; r < 8; ++r)
      acc[r] = Cinit[(long long)bz * sC + (long long)(row0 + r) * ldc + col];
  } else {
#pragma unroll
    for (int r = 0; r < 8; ++r) acc[r] = 0.0f;
  }

  for (int k0 = 0; k0 < K; k0 += 32) {
    if (k0 + 64 < K) __builtin_prefetch((const void*)(Ab + k0 + 64), 0, 1);
    acc = wmma16(load_a_frag(Ab + k0, lda),
                 load_b_packed(Bb, k0 >> 5, tn, ntiles), acc);
  }

  // epilogue: uniform guards hoisted out of the unrolled loops
  float vals[8];
  const float bv = bias ? bias[col] : 0.0f;
#pragma unroll
  for (int r = 0; r < 8; ++r) vals[r] = acc[r] + bv;
  if (gelu) {
#pragma unroll
    for (int r = 0; r < 8; ++r) vals[r] = gelu_exact(vals[r]);
  }
  const long long base = (long long)bz * sC + (long long)row0 * ldc + col;
  if (Cf32) {
#pragma unroll
    for (int r = 0; r < 8; ++r) Cf32[base + (long long)r * ldc] = vals[r];
  }
  if (Cf16) {
#pragma unroll
    for (int r = 0; r < 8; ++r) Cf16[base + (long long)r * ldc] = (_Float16)vals[r];
  }
  if (CT) {  // rows are (b, n) of [B*NXX]; store [b][col][n]
#pragma unroll
    for (int r = 0; r < 8; ++r) {
      const int row = row0 + r;
      CT[((long long)(row >> 11) * DMODEL + col) * NXX + (row & (NXX - 1))] =
          (_Float16)vals[r];
    }
  }
}

// ---------------------------------------------------------------------------
// Spectral mode mixing: Or = Xr*Wr - Xi*Wi ; Oi = Xr*Wi + Xi*Wr
// grid = (4 out-ch tiles, 16 modes); block = 32. W slices fragment-packed.
// ---------------------------------------------------------------------------
__global__ __launch_bounds__(32)
void spectral_mul(const _Float16* __restrict__ modes16,
                  const _Float16* __restrict__ Wr, const _Float16* __restrict__ Wi,
                  const _Float16* __restrict__ Win, _Float16* __restrict__ out16) {
  const int tn = blockIdx.x;
  const int k = blockIdx.y;
  const int lane = threadIdx.x & 31;
  const int hi = lane >> 4, nn = lane & 15;
  const int bstride = 2 * NMODE * WID;

  const _Float16* Xr = modes16 + k * WID;
  const _Float16* Xi = modes16 + (NMODE + k) * WID;
  const _Float16* Br = Wr + (long long)k * WID * WID;
  const _Float16* Bi = Wi + (long long)k * WID * WID;
  const _Float16* Bn = Win + (long long)k * WID * WID;

  v8f accR, accI;
#pragma unroll
  for (int r = 0; r < 8; ++r) { accR[r] = 0.0f; accI[r] = 0.0f; }

#pragma unroll
  for (int k0 = 0; k0 < WID; k0 += 32) {
    const int kt = k0 >> 5;
    v16h ar = load_a_frag(Xr + k0, bstride);
    v16h ai = load_a_frag(Xi + k0, bstride);
    accR = wmma16(ar, load_b_packed(Br, kt, tn, 4), accR);
    accR = wmma16(ai, load_b_packed(Bn, kt, tn, 4), accR);
    accI = wmma16(ar, load_b_packed(Bi, kt, tn, 4), accI);
    accI = wmma16(ai, load_b_packed(Br, kt, tn, 4), accI);
  }
#pragma unroll
  for (int r = 0; r < 8; ++r) {
    const int b = hi * 8 + r;
    out16[(long long)b * bstride + k * WID + tn * 16 + nn] = (_Float16)accR[r];
    out16[(long long)b * bstride + (NMODE + k) * WID + tn * 16 + nn] =
        (_Float16)accI[r];
  }
}

// ---------------------------------------------------------------------------
// Flash attention with double-buffered TDM staging: 4 waves / block share one
// batch; wave 0 issues next chunk's K (32x128) and V^T (128x32) TDM loads
// BEFORE computing the current chunk, so the tensor DMA overlaps the 16 WMMAs
// + online softmax. s_wait_tensorcnt only at the chunk boundary.
// grid = 512, block = 128.
// ---------------------------------------------------------------------------
__global__ __launch_bounds__(128)
void flash_attn(const _Float16* __restrict__ Q, const _Float16* __restrict__ Km,
                const _Float16* __restrict__ Vt, _Float16* __restrict__ O) {
  __shared__ __align__(16) _Float16 ktile[2][32 * DMODEL];  // [key][d]
  __shared__ __align__(16) _Float16 vtile[2][DMODEL * 32];  // [d][key]
  __shared__ __align__(16) _Float16 pbuf[4][16 * 32];
  const int wave = threadIdx.x >> 5;
  const int lane = threadIdx.x & 31;
  const int hi = lane >> 4, nn = lane & 15;
  const int qt = blockIdx.x * 4 + wave;  // all 4 waves: same batch (4 | 128)
  const int b = qt >> 7;
  const int q0 = (qt & 127) * 16;
  const _Float16* Qb = Q + ((long long)b * NXX + q0) * DMODEL;
  const _Float16* Kb = Km + (long long)b * NXX * DMODEL;
  const _Float16* Vb = Vt + (long long)b * DMODEL * NXX;
  _Float16* pb = pbuf[wave];
  const unsigned klds0 = (unsigned)(uintptr_t)&ktile[0][0];
  const unsigned klds1 = (unsigned)(uintptr_t)&ktile[1][0];
  const unsigned vlds0 = (unsigned)(uintptr_t)&vtile[0][0];
  const unsigned vlds1 = (unsigned)(uintptr_t)&vtile[1][0];

  v16h qf[4];
#pragma unroll
  for (int c = 0; c < 4; ++c) qf[c] = load_a_frag(Qb + c * 32, DMODEL);

  v8f oacc[8];
#pragma unroll
  for (int t = 0; t < 8; ++t)
#pragma unroll
    for (int r = 0; r < 8; ++r) oacc[t][r] = 0.0f;
  float mrow[8], lrow[8];
#pragma unroll
  for (int r = 0; r < 8; ++r) { mrow[r] = -3.0e38f; lrow[r] = 0.0f; }

  // prologue: stage chunk 0 into buffer 0
  if (wave == 0) {
    tdm_load_2d(klds0, Kb, DMODEL, 32, DMODEL);
    tdm_load_2d(vlds0, Vb, 32, DMODEL, NXX);
    __builtin_amdgcn_s_wait_tensorcnt(0);
  }
  __syncthreads();

  int cur = 0;
  for (int j = 0; j < NXX; j += 32) {
    // kick off next chunk's DMA into the other buffer (overlaps compute)
    if (wave == 0 && j + 32 < NXX) {
      tdm_load_2d(cur ? klds0 : klds1, Kb + (long long)(j + 32) * DMODEL,
                  DMODEL, 32, DMODEL);
      tdm_load_2d(cur ? vlds0 : vlds1, Vb + j + 32, 32, DMODEL, NXX);
    }
    const _Float16* kt_ = ktile[cur];
    const _Float16* vt_ = vtile[cur];

    v8f s0, s1;
#pragma unroll
    for (int r = 0; r < 8; ++r) { s0[r] = 0.0f; s1[r] = 0.0f; }
#pragma unroll
    for (int c = 0; c < 4; ++c) {
      s0 = wmma16(qf[c], load_b_t(kt_ + c * 32, DMODEL), s0);
      s1 = wmma16(qf[c], load_b_t(kt_ + 16 * DMODEL + c * 32, DMODEL), s1);
    }
    float alpha[8];
#pragma unroll
    for (int r = 0; r < 8; ++r) {
      float mloc = fmaxf(s0[r], s1[r]);
#pragma unroll
      for (int off = 1; off < 16; off <<= 1)
        mloc = fmaxf(mloc, __shfl_xor(mloc, off, 32));
      const float mnew = fmaxf(mrow[r], mloc);
      alpha[r] = __expf(mrow[r] - mnew);
      mrow[r] = mnew;
      const float p0 = __expf(s0[r] - mnew);
      const float p1 = __expf(s1[r] - mnew);
      float ls = p0 + p1;
#pragma unroll
      for (int off = 1; off < 16; off <<= 1) ls += __shfl_xor(ls, off, 32);
      lrow[r] = lrow[r] * alpha[r] + ls;
      const int row = hi * 8 + r;
      pb[row * 32 + nn] = (_Float16)p0;
      pb[row * 32 + nn + 16] = (_Float16)p1;
    }
#pragma unroll
    for (int t = 0; t < 8; ++t)
#pragma unroll
      for (int r = 0; r < 8; ++r) oacc[t][r] *= alpha[r];

    v16h pf = load_a_frag(pb, 32);  // per-wave buffer; DS ops in-order per wave
#pragma unroll
    for (int t = 0; t < 8; ++t)
      oacc[t] = wmma16(pf, load_b_t(vt_ + t * 16 * 32, 32), oacc[t]);

    if (wave == 0) __builtin_amdgcn_s_wait_tensorcnt(0);
    __syncthreads();  // next buffer ready + this buffer fully consumed
    cur ^= 1;
  }
#pragma unroll
  for (int t = 0; t < 8; ++t)
#pragma unroll
    for (int r = 0; r < 8; ++r) {
      const float v = oacc[t][r] / lrow[r];
      O[((long long)b * NXX + q0 + hi * 8 + r) * DMODEL + t * 16 + nn] =
          (_Float16)v;
    }
}

// ---------------------------------------------------------------------------
// Elementwise / prep kernels
// ---------------------------------------------------------------------------
__global__ void fc0_kernel(const float* __restrict__ x, const float* __restrict__ g,
                           const float* __restrict__ w, const float* __restrict__ bias,
                           _Float16* __restrict__ h16) {
  const int idx = blockIdx.x * 256 + threadIdx.x;
  const int wo = idx & 63;
  const int n = (idx >> 6) & (NXX - 1);
  const int b = idx >> 17;
  const float v = x[b * NXX + n] * w[wo] + g[n] * w[WID + wo] + bias[wo];
  h16[idx] = (_Float16)v;
}

__global__ void prep_basis(_Float16* __restrict__ Bf, _Float16* __restrict__ Bi) {
  const int idx = blockIdx.x * 256 + threadIdx.x;  // 32*2048
  const int n = idx & (NXX - 1);
  const int row = idx >> 11;
  const int k = row & 15;
  const int t = (k * n) & (NXX - 1);
  const float ang = 6.283185307179586f * (float)t * (1.0f / NXX);
  const float c = cosf(ang), s = sinf(ang);
  Bf[row * NXX + n] = (_Float16)((row < NMODE) ? c : -s);
  const float ck = (k == 0) ? (1.0f / NXX) : (2.0f / NXX);
  Bi[n * 32 + row] = (_Float16)((row < NMODE) ? ck * c : -ck * s);
}

// repack a row-major f16 activation [K, N] (per batch) into fragment order
__global__ void pack_act(const _Float16* __restrict__ s, _Float16* __restrict__ d,
                         int K, int N, long long sIn, long long sOut) {
  const int idx = blockIdx.x * 256 + threadIdx.x;
  const int bz = blockIdx.y;
  if (idx < K * N) {
    const int k = idx / N, n = idx - k * N;
    d[(long long)bz * sOut + fragidx(k, n, N >> 4)] = s[(long long)bz * sIn + idx];
  }
}

__global__ void pack_spectral(const float* __restrict__ wr, const float* __restrict__ wi,
                              _Float16* __restrict__ Wr, _Float16* __restrict__ Wi,
                              _Float16* __restrict__ Win) {
  const int idx = blockIdx.x * 256 + threadIdx.x;  // 3*64*64*16
  const int k = idx & 15;
  const int o = (idx >> 4) & 63;
  const int i = (idx >> 10) & 63;
  const int blk = idx >> 16;
  const long long dst =
      (long long)(blk * NMODE + k) * (WID * WID) + fragidx(i, o, 4);
  const float r = wr[idx], m = wi[idx];
  Wr[dst] = (_Float16)r;
  Wi[dst] = (_Float16)m;
  Win[dst] = (_Float16)(-m);
}

__global__ void pack_conv(const float* __restrict__ w_w, _Float16* __restrict__ WT) {
  const int idx = blockIdx.x * 256 + threadIdx.x;  // 3*64*64
  const int o = idx & 63;
  const int i = (idx >> 6) & 63;
  const int blk = idx >> 12;
  WT[(long long)blk * (WID * WID) + fragidx(i, o, 4)] =
      (_Float16)w_w[blk * 4096 + o * 64 + i];
}

__global__ void pack_cvt(const float* __restrict__ s, _Float16* __restrict__ d,
                         int K, int N) {
  const int idx = blockIdx.x * 256 + threadIdx.x;
  if (idx < K * N) d[fragidx(idx / N, idx % N, N >> 4)] = (_Float16)s[idx];
}

__global__ void pack_qkv(const float* __restrict__ qkv_w, _Float16* __restrict__ Qw,
                         _Float16* __restrict__ Kw, _Float16* __restrict__ Vw) {
  const int idx = blockIdx.x * 256 + threadIdx.x;  // 128*128
  const int d = idx & 127;
  const int i = idx >> 7;
  const long long dst = fragidx(i, d, 8);
  const float sc = 0.08838834764831845f;  // DM^-0.5 folded into Q weights
  Qw[dst] = (_Float16)(qkv_w[i * 384 + 3 * d + 0] * sc);
  Kw[dst] = (_Float16)(qkv_w[i * 384 + 3 * d + 1]);
  Vw[dst] = (_Float16)(qkv_w[i * 384 + 3 * d + 2]);
}

__global__ __launch_bounds__(256)
void mean_kernel(const float* __restrict__ o2, float* __restrict__ out) {
  __shared__ float red[256];
  const int b = blockIdx.x >> 7;
  const int d = blockIdx.x & 127;
  float s = 0.0f;
  for (int n = threadIdx.x; n < NXX; n += 256)
    s += o2[((long long)b * NXX + n) * DMODEL + d];
  red[threadIdx.x] = s;
  __syncthreads();
  for (int st = 128; st > 0; st >>= 1) {
    if (threadIdx.x < st) red[threadIdx.x] += red[threadIdx.x + st];
    __syncthreads();
  }
  if (threadIdx.x == 0) out[blockIdx.x] = red[0] * (1.0f / NXX);
}

// ---------------------------------------------------------------------------
extern "C" void kernel_launch(void* const* d_in, const int* in_sizes, int n_in,
                              void* d_out, int out_size, void* d_ws, size_t ws_size,
                              hipStream_t stream) {
  (void)in_sizes; (void)n_in; (void)out_size; (void)ws_size;
  const float* x      = (const float*)d_in[0];
  const float* gridv  = (const float*)d_in[1];
  const float* fc0_w  = (const float*)d_in[2];
  const float* fc0_b  = (const float*)d_in[3];
  const float* sc_wr  = (const float*)d_in[4];
  const float* sc_wi  = (const float*)d_in[5];
  const float* w_w    = (const float*)d_in[6];
  const float* w_b    = (const float*)d_in[7];
  const float* fc1_w  = (const float*)d_in[8];
  const float* fc1_b  = (const float*)d_in[9];
  const float* qkv_w  = (const float*)d_in[10];
  const float* lin_w1 = (const float*)d_in[11];
  const float* lin_b1 = (const float*)d_in[12];
  const float* lin_w2 = (const float*)d_in[13];
  const float* lin_b2 = (const float*)d_in[14];

  char* base = (char*)d_ws;
  size_t off = 0;
  auto alloc = [&](size_t bytes) -> char* {
    char* r = base + off;
    off += (bytes + 255) & ~(size_t)255;
    return r;
  };

  const size_t NHW = (size_t)BATCH * NXX * WID;
  const size_t NHD = (size_t)BATCH * NXX * DMODEL;

  _Float16* hA     = (_Float16*)alloc(NHW * 2);
  _Float16* hB     = (_Float16*)alloc(NHW * 2);
  _Float16* hPack  = (_Float16*)alloc(NHW * 2);  // h in fragment order (DFT B)
  _Float16* Bf     = (_Float16*)alloc((size_t)32 * NXX * 2);
  _Float16* Bi     = (_Float16*)alloc((size_t)NXX * 32 * 2);
  _Float16* modes  = (_Float16*)alloc((size_t)BATCH * 32 * WID * 2);
  _Float16* modesO = (_Float16*)alloc((size_t)BATCH * 32 * WID * 2);
  _Float16* mPack  = (_Float16*)alloc((size_t)BATCH * 32 * WID * 2);
  _Float16* Wr_p   = (_Float16*)alloc((size_t)3 * NMODE * WID * WID * 2);
  _Float16* Wi_p   = (_Float16*)alloc((size_t)3 * NMODE * WID * WID * 2);
  _Float16* Win_p  = (_Float16*)alloc((size_t)3 * NMODE * WID * WID * 2);
  _Float16* WT     = (_Float16*)alloc((size_t)3 * WID * WID * 2);
  _Float16* fc1w16 = (_Float16*)alloc((size_t)WID * DMODEL * 2);
  _Float16* Qw16   = (_Float16*)alloc((size_t)DMODEL * DMODEL * 2);
  _Float16* Kw16   = (_Float16*)alloc((size_t)DMODEL * DMODEL * 2);
  _Float16* Vw16   = (_Float16*)alloc((size_t)DMODEL * DMODEL * 2);
  _Float16* w116   = (_Float16*)alloc((size_t)DMODEL * DMODEL * 2);
  _Float16* w216   = (_Float16*)alloc((size_t)DMODEL * DMODEL * 2);
  float*    x1     = (float*)alloc(NHW * 4);
  _Float16* g16    = (_Float16*)alloc(NHD * 2);
  _Float16* Qf     = (_Float16*)alloc(NHD * 2);
  _Float16* Kf     = (_Float16*)alloc(NHD * 2);
  _Float16* Vtr    = (_Float16*)alloc(NHD * 2);  // V transposed [B][DM][NX]
  _Float16* Of     = (_Float16*)alloc(NHD * 2);
  _Float16* u16    = (_Float16*)alloc(NHD * 2);
  float*    o2     = (float*)alloc(NHD * 4);

  const dim3 wblk(32);
  const long long sHW = (long long)NXX * WID;   // per-batch [2048,64]
  const long long sM  = 32LL * WID;             // per-batch [32,64]

  // --- weight / basis prep (all weight B matrices -> fragment-linear f16) ---
  prep_basis<<<256, 256, 0, stream>>>(Bf, Bi);
  pack_spectral<<<768, 256, 0, stream>>>(sc_wr, sc_wi, Wr_p, Wi_p, Win_p);
  pack_conv<<<48, 256, 0, stream>>>(w_w, WT);
  pack_cvt<<<32, 256, 0, stream>>>(fc1_w, fc1w16, WID, DMODEL);
  pack_qkv<<<64, 256, 0, stream>>>(qkv_w, Qw16, Kw16, Vw16);
  pack_cvt<<<64, 256, 0, stream>>>(lin_w1, w116, DMODEL, DMODEL);
  pack_cvt<<<64, 256, 0, stream>>>(lin_w2, w216, DMODEL, DMODEL);

  // --- lift ---
  fc0_kernel<<<8192, 256, 0, stream>>>(x, gridv, fc0_w, fc0_b, hA);

  // --- 3 Fourier blocks ---
  _Float16* cur = hA;
  _Float16* nxt = hB;
  for (int blk = 0; blk < 3; ++blk) {
    // fragment-pack h for use as DFT B
    pack_act<<<dim3(512, BATCH), 256, 0, stream>>>(cur, hPack, NXX, WID, sHW, sHW);
    // truncated DFT: modes[b][32][64] = Bf[32,2048] @ h[b][2048,64]
    wmma_gemm<<<dim3(4, 2, 16), wblk, 0, stream>>>(
        Bf, hPack, nullptr, nullptr, modes, nullptr, nullptr, 0,
        NXX, NXX, WID, 0LL, sHW, sM);
    // complex channel mixing per mode
    spectral_mul<<<dim3(4, 16), wblk, 0, stream>>>(
        modes, Wr_p + (size_t)blk * NMODE * WID * WID,
        Wi_p + (size_t)blk * NMODE * WID * WID,
        Win_p + (size_t)blk * NMODE * WID * WID, modesO);
    // fragment-pack modesO for use as iDFT B
    pack_act<<<dim3(8, BATCH), 256, 0, stream>>>(modesO, mPack, 32, WID, sM, sM);
    // inverse DFT: x1[b][2048][64] = Bi[2048,32] @ modesO[b][32,64]
    wmma_gemm<<<dim3(4, 128, 16), wblk, 0, stream>>>(
        Bi, mPack, nullptr, x1, nullptr, nullptr, nullptr, 0,
        32, 32, WID, 0LL, sM, sHW);
    // 1x1 conv + spectral part + bias + gelu (fused epilogue)
    wmma_gemm<<<dim3(4, 2048, 1), wblk, 0, stream>>>(
        cur, WT + (size_t)blk * WID * WID, x1, nullptr, nxt, nullptr,
        w_b + blk * WID, 1, WID, WID, WID, 0LL, 0LL, 0LL);
    _Float16* tmp = cur; cur = nxt; nxt = tmp;
  }

  // --- fc1 ---
  wmma_gemm<<<dim3(8, 2048, 1), wblk, 0, stream>>>(
      cur, fc1w16, nullptr, nullptr, g16, nullptr, fc1_b, 0,
      WID, WID, DMODEL, 0LL, 0LL, 0LL);

  // --- Q, K, V projections (interleave + softmax scale folded into weights) ---
  wmma_gemm<<<dim3(8, 2048, 1), wblk, 0, stream>>>(
      g16, Qw16, nullptr, nullptr, Qf, nullptr, nullptr, 0,
      DMODEL, DMODEL, DMODEL, 0LL, 0LL, 0LL);
  wmma_gemm<<<dim3(8, 2048, 1), wblk, 0, stream>>>(
      g16, Kw16, nullptr, nullptr, Kf, nullptr, nullptr, 0,
      DMODEL, DMODEL, DMODEL, 0LL, 0LL, 0LL);
  wmma_gemm<<<dim3(8, 2048, 1), wblk, 0, stream>>>(
      g16, Vw16, nullptr, nullptr, nullptr, Vtr, nullptr, 0,
      DMODEL, DMODEL, DMODEL, 0LL, 0LL, 0LL);

  // --- flash attention (double-buffered TDM K / V^T tiles in LDS) ---
  flash_attn<<<512, 128, 0, stream>>>(Qf, Kf, Vtr, Of);

  // --- littleFNN ---
  wmma_gemm<<<dim3(8, 2048, 1), wblk, 0, stream>>>(
      Of, w116, nullptr, nullptr, u16, nullptr, lin_b1, 1,
      DMODEL, DMODEL, DMODEL, 0LL, 0LL, 0LL);
  wmma_gemm<<<dim3(8, 2048, 1), wblk, 0, stream>>>(
      u16, w216, nullptr, o2, nullptr, nullptr, lin_b2, 0,
      DMODEL, DMODEL, DMODEL, 0LL, 0LL, 0LL);

  // --- mean over sequence ---
  mean_kernel<<<2048, 256, 0, stream>>>(o2, (float*)d_out);
}
